// SubsetSelector_6889127543098
// MI455X (gfx1250) — matrix operand order repaired and checked by reading, not compiled, not run on hardware
//
#include <hip/hip_runtime.h>
#include <math.h>

// Problem constants (from reference): B=256, H=512, D=64, K=128, DEPTH=16
constexpr int NB     = 256;
constexpr int NH     = 512;
constexpr int ND     = 64;
constexpr int NK     = 128;
constexpr int NDEPTH = 16;
constexpr float TEMPR = 20.0f;
constexpr float NEGV  = -1.0e9f;

constexpr int PITCH = 520; // bf16 row pitch for LDS state (16B aligned, staggers banks)

typedef unsigned short u16;
typedef __attribute__((ext_vector_type(16))) __bf16 v16bf;
typedef __attribute__((ext_vector_type(8)))  float  v8f;
typedef __attribute__((ext_vector_type(8)))  unsigned short u16x8;

union Frag {
  u16   s[16];
  u16x8 h[2];
  v16bf v;
};

__device__ __forceinline__ u16 f32_to_bf16(float f) {
  union { float f; unsigned u; } c; c.f = f;
  unsigned u = c.u;
  u += 0x7FFFu + ((u >> 16) & 1u);   // round-to-nearest-even
  return (u16)(u >> 16);
}

// lexicographic better-than: value desc, index asc (matches top_k ordering)
__device__ __forceinline__ bool tk_gt(float a, int ai, float b, int bi) {
  return (a > b) || (a == b && ai < bi);
}

// insert (x,ix) into sorted-descending 4-list held in registers
__device__ __forceinline__ void tk_ins4(float& v0, float& v1, float& v2, float& v3,
                                        int& i0, int& i1, int& i2, int& i3,
                                        float x, int ix) {
  if (tk_gt(x, ix, v3, i3)) {
    v3 = x; i3 = ix;
    if (tk_gt(v3, i3, v2, i2)) {
      float tv = v2; int ti = i2; v2 = v3; i2 = i3; v3 = tv; i3 = ti;
      if (tk_gt(v2, i2, v1, i1)) {
        tv = v1; ti = i1; v1 = v2; i1 = i2; v2 = tv; i2 = ti;
        if (tk_gt(v1, i1, v0, i0)) {
          tv = v0; ti = i0; v0 = v1; i0 = i1; v1 = tv; i1 = ti;
        }
      }
    }
  }
}

// ---------------------------------------------------------------------------
// Prep: fp32 weight [Hk x ncols] -> bf16 WMMA B-fragment layout:
//   frag[((kc*ntiles + nT)*32 + lane)*16 + e]
//   element e of lane l = W[(kc*32 + ((l&16)>>1) + (e&7) + ((e&8)<<1)) * ncols + nT*16 + (l&15)]
// ---------------------------------------------------------------------------
__global__ void SubsetSelector_prep_kernel(const float* __restrict__ W,
                                           u16* __restrict__ frag,
                                           int ncols, int ntiles, int total) {
  int idx = blockIdx.x * blockDim.x + threadIdx.x;
  if (idx >= total) return;
  int e    = idx & 15;
  int lane = (idx >> 4) & 31;
  int rest = idx >> 9;
  int nT   = rest % ntiles;
  int kc   = rest / ntiles;
  int k = kc * 32 + ((lane & 16) >> 1) + (e & 7) + ((e & 8) << 1);
  int n = nT * 16 + (lane & 15);
  frag[idx] = f32_to_bf16(W[k * ncols + n]);
}

// ---------------------------------------------------------------------------
// Main: one workgroup per batch row; full 16-step beam search resident in LDS.
// 512 threads = 16 wave32s.
// ---------------------------------------------------------------------------

// LDS layout (bytes)
constexpr int OFF_STATE0 = 0;                                     // 128*520*2 = 133120
constexpr int OFF_STATE1 = OFF_STATE0 + NK * PITCH * 2;           // 133120
constexpr int OFF_CAND   = OFF_STATE1 + NK * PITCH * 2;           // 266240, 8192 f32
constexpr int OFF_SCORES = OFF_CAND   + NK * ND * 4;              // 299008
constexpr int OFF_SCOR2  = OFF_SCORES + NK * 4;                   // 299520
constexpr int OFF_HIST   = OFF_SCOR2  + NK * 4;                   // 300032, 128*16 i32
constexpr int OFF_BEAM   = OFF_HIST   + NK * NDEPTH * 4;          // 308224
constexpr int OFF_DEC    = OFF_BEAM   + NK * 4;                   // 308736
constexpr int OFF_WVAL   = OFF_DEC    + NK * 4;                   // 309248, 16 waves * 4
constexpr int OFF_WIDX   = OFF_WVAL   + 64 * 4;                   // 309504
constexpr int SMEM_BYTES = OFF_WIDX   + 64 * 4;                   // 309760

__global__ __launch_bounds__(512)
void SubsetSelector_6889127543098_kernel(const u16* __restrict__ wlFrag,
                                         const u16* __restrict__ whhFrag,
                                         const float* __restrict__ b_logits,
                                         const float* __restrict__ b_hh,
                                         const float* __restrict__ emb,
                                         const float* __restrict__ root,
                                         int* __restrict__ outHist,
                                         float* __restrict__ outScores) {
  extern __shared__ char smem[];
  u16*   state0  = (u16*)(smem + OFF_STATE0);
  u16*   state1  = (u16*)(smem + OFF_STATE1);
  float* cand    = (float*)(smem + OFF_CAND);
  float* scores  = (float*)(smem + OFF_SCORES);
  float* scores2 = (float*)(smem + OFF_SCOR2);
  int*   hist    = (int*)(smem + OFF_HIST);
  int*   beamIdx = (int*)(smem + OFF_BEAM);
  int*   decs    = (int*)(smem + OFF_DEC);
  float* wval    = (float*)(smem + OFF_WVAL);
  int*   widx    = (int*)(smem + OFF_WIDX);

  const int tid  = threadIdx.x;
  const int lane = tid & 31;
  const int wave = tid >> 5;
  const int b    = blockIdx.x;

  // --- init: all beams at root state; only beam 0 live; hist zero
  for (int i = tid; i < NK * NH; i += 512) {
    int r = i >> 9, h = i & (NH - 1);
    state0[r * PITCH + h] = f32_to_bf16(root[h]);
  }
  if (tid < NK) scores[tid] = (tid == 0) ? 0.0f : NEGV;
  for (int i = tid; i < NK * NDEPTH; i += 512) hist[i] = 0;
  __syncthreads();

  u16* cur = state0;
  u16* nxt = state1;

  const int mOff = (lane >> 4) << 3; // 0 (lanes 0-15) or 8 (lanes 16-31)
  const int nSub = lane & 15;

  // Per-wave tiling: GEMM1 mT = wave>>1, nT in { (wave&1)*2 + j }, j=0..1.
  //                  GEMM2 mT = wave>>1, nT in { (wave&1)*16 + j }, j=0..15.
  const int mT    = wave >> 1;
  const int nHalf = wave & 1;

  for (int t = 0; t < NDEPTH; ++t) {
    // ======== GEMM1: logits[128,64] = cur[128,512] @ Wl ; into cand ========
    {
      const int arow = (mT * 16 + nSub) * PITCH;
      const int nT0  = nHalf * 2;
      v8f acc0 = {}, acc1 = {};
      #pragma unroll 2
      for (int kc = 0; kc < 16; ++kc) {
        Frag a, b0, b1;
        const u16* ap = cur + arow + kc * 32 + mOff;
        a.h[0] = *(const u16x8*)(ap);        // K = kc*32 + mOff + 0..7
        a.h[1] = *(const u16x8*)(ap + 16);   // K = kc*32 + mOff + 16..23
        const u16* bp = wlFrag + (((kc << 2) + nT0) * 32 + lane) * 16;
        b0.h[0] = *(const u16x8*)(bp);
        b0.h[1] = *(const u16x8*)(bp + 8);
        b1.h[0] = *(const u16x8*)(bp + 512);     // next nT fragment (+32*16)
        b1.h[1] = *(const u16x8*)(bp + 512 + 8);
        acc0 = __builtin_amdgcn_wmma_f32_16x16x32_bf16(
                   false, a.v, false, b0.v, (short)0, acc0, false, false);
        acc1 = __builtin_amdgcn_wmma_f32_16x16x32_bf16(
                   false, a.v, false, b1.v, (short)0, acc1, false, false);
      }
      const int mb = mT * 16 + mOff;
      {
        const int n = nT0 * 16 + nSub;
        const float bl = b_logits[n];
        #pragma unroll
        for (int v = 0; v < 8; ++v) cand[(mb + v) * ND + n] = acc0[v] + bl;
      }
      {
        const int n = (nT0 + 1) * 16 + nSub;
        const float bl = b_logits[n];
        #pragma unroll
        for (int v = 0; v < 8; ++v) cand[(mb + v) * ND + n] = acc1[v] + bl;
      }
    }
    __syncthreads();

    // ======== temperature log-softmax + cumulative scores ========
    if (tid < NK) {
      const float inv = 1.0f / TEMPR;
      float mx = -3.4e38f;
      for (int d = 0; d < ND; ++d) mx = fmaxf(mx, cand[tid * ND + d] * inv);
      float s = 0.0f;
      for (int d = 0; d < ND; ++d) s += expf(cand[tid * ND + d] * inv - mx);
      const float lz = mx + logf(s);
      const float sc = scores[tid];
      for (int d = 0; d < ND; ++d)
        cand[tid * ND + d] = sc + (cand[tid * ND + d] * inv - lz);
    }
    __syncthreads();

    // ======== top-128 of 8192: 32 rounds extracting global top-4 ========
    // Round = (per-thread sorted top-4 of 16 strided cands) -> (butterfly
    // shuffle merge of disjoint sorted lists -> wave top-4) -> (wave 0 merges
    // 16x4 partials -> global top-4) -> commit + mask. 2 barriers per round.
    for (int it = 0; it < NK; it += 4) {
      float v0 = -3.4e38f, v1 = -3.4e38f, v2 = -3.4e38f, v3 = -3.4e38f;
      int   i0 = 0x7fffffff, i1 = 0x7fffffff, i2 = 0x7fffffff, i3 = 0x7fffffff;
      for (int j = tid; j < NK * ND; j += 512)
        tk_ins4(v0, v1, v2, v3, i0, i1, i2, i3, cand[j], j);
      #pragma unroll
      for (int off = 16; off > 0; off >>= 1) {
        float w0 = __shfl_xor(v0, off, 32), w1 = __shfl_xor(v1, off, 32);
        float w2 = __shfl_xor(v2, off, 32), w3 = __shfl_xor(v3, off, 32);
        int   k0 = __shfl_xor(i0, off, 32), k1 = __shfl_xor(i1, off, 32);
        int   k2 = __shfl_xor(i2, off, 32), k3 = __shfl_xor(i3, off, 32);
        tk_ins4(v0, v1, v2, v3, i0, i1, i2, i3, w0, k0);
        tk_ins4(v0, v1, v2, v3, i0, i1, i2, i3, w1, k1);
        tk_ins4(v0, v1, v2, v3, i0, i1, i2, i3, w2, k2);
        tk_ins4(v0, v1, v2, v3, i0, i1, i2, i3, w3, k3);
      }
      if (lane == 0) {
        wval[wave * 4 + 0] = v0; widx[wave * 4 + 0] = i0;
        wval[wave * 4 + 1] = v1; widx[wave * 4 + 1] = i1;
        wval[wave * 4 + 2] = v2; widx[wave * 4 + 2] = i2;
        wval[wave * 4 + 3] = v3; widx[wave * 4 + 3] = i3;
      }
      __syncthreads();
      if (wave == 0) {
        // 64 partial entries, 2 per lane (each wave's list is sorted desc)
        float a0 = wval[lane * 2], a1 = wval[lane * 2 + 1];
        int   c0 = widx[lane * 2], c1 = widx[lane * 2 + 1];
        v0 = a0; i0 = c0; v1 = a1; i1 = c1;
        v2 = -3.4e38f; i2 = 0x7fffffff; v3 = -3.4e38f; i3 = 0x7fffffff;
        if (tk_gt(v1, i1, v0, i0)) {
          float tv = v0; int ti = i0; v0 = v1; i0 = i1; v1 = tv; i1 = ti;
        }
        #pragma unroll
        for (int off = 16; off > 0; off >>= 1) {
          float w0 = __shfl_xor(v0, off, 32), w1 = __shfl_xor(v1, off, 32);
          float w2 = __shfl_xor(v2, off, 32), w3 = __shfl_xor(v3, off, 32);
          int   k0 = __shfl_xor(i0, off, 32), k1 = __shfl_xor(i1, off, 32);
          int   k2 = __shfl_xor(i2, off, 32), k3 = __shfl_xor(i3, off, 32);
          tk_ins4(v0, v1, v2, v3, i0, i1, i2, i3, w0, k0);
          tk_ins4(v0, v1, v2, v3, i0, i1, i2, i3, w1, k1);
          tk_ins4(v0, v1, v2, v3, i0, i1, i2, i3, w2, k2);
          tk_ins4(v0, v1, v2, v3, i0, i1, i2, i3, w3, k3);
        }
        if (lane == 0) {
          scores2[it + 0] = v0; beamIdx[it + 0] = i0 >> 6; decs[it + 0] = i0 & 63;
          scores2[it + 1] = v1; beamIdx[it + 1] = i1 >> 6; decs[it + 1] = i1 & 63;
          scores2[it + 2] = v2; beamIdx[it + 2] = i2 >> 6; decs[it + 2] = i2 & 63;
          scores2[it + 3] = v3; beamIdx[it + 3] = i3 >> 6; decs[it + 3] = i3 & 63;
          cand[i0] = -3.4e38f; cand[i1] = -3.4e38f;
          cand[i2] = -3.4e38f; cand[i3] = -3.4e38f;
        }
      }
      __syncthreads();
    }

    // ======== history gather + score commit (register ping-pong) ========
    int hreg[NDEPTH];
    int par = (tid < NK) ? beamIdx[tid] : 0;
    if (tid < NK) {
      #pragma unroll
      for (int j = 0; j < NDEPTH; ++j) hreg[j] = hist[par * NDEPTH + j];
    }
    __syncthreads();
    if (tid < NK) {
      #pragma unroll
      for (int j = 0; j < NDEPTH; ++j) hist[tid * NDEPTH + j] = hreg[j];
      hist[tid * NDEPTH + t] = decs[tid];
      scores[tid] = scores2[tid];
    }
    __syncthreads();

    // ======== GEMM2: nxt = tanh(cur[beamIdx] @ Whh + b_hh + emb[dec]) ======
    // Each wave: fixed mT, 16 nTiles in two passes of 8 accumulators.
    {
      const int arow = beamIdx[mT * 16 + nSub] * PITCH;  // parent-state gather
      const int mb   = mT * 16 + mOff;
      int dd[8];
      #pragma unroll
      for (int v = 0; v < 8; ++v) dd[v] = decs[mb + v];

      for (int pass = 0; pass < 2; ++pass) {
        const int nT0 = nHalf * 16 + pass * 8;
        v8f acc[8];
        #pragma unroll
        for (int j = 0; j < 8; ++j) acc[j] = (v8f){};

        #pragma unroll 2
        for (int kc = 0; kc < 16; ++kc) {
          Frag a;
          const u16* ap = cur + arow + kc * 32 + mOff;
          a.h[0] = *(const u16x8*)(ap);
          a.h[1] = *(const u16x8*)(ap + 16);
          const u16* bbase = whhFrag + (((kc << 5) + nT0) * 32 + lane) * 16;
          #pragma unroll
          for (int j = 0; j < 8; ++j) {
            Frag bm;
            const u16* bp = bbase + j * 512;   // +32 lanes * 16 elems per nT
            bm.h[0] = *(const u16x8*)(bp);
            bm.h[1] = *(const u16x8*)(bp + 8);
            acc[j] = __builtin_amdgcn_wmma_f32_16x16x32_bf16(
                         false, a.v, false, bm.v, (short)0, acc[j], false, false);
          }
        }

        #pragma unroll
        for (int j = 0; j < 8; ++j) {
          const int n  = (nT0 + j) * 16 + nSub;
          const float bb = b_hh[n];
          #pragma unroll
          for (int v = 0; v < 8; ++v) {
            float val = acc[j][v] + bb + emb[dd[v] * NH + n];
            nxt[(mb + v) * PITCH + n] = f32_to_bf16(tanhf(val));
          }
        }
      }
    }
    __syncthreads();
    u16* tmp = cur; cur = nxt; nxt = tmp;
  }

  // ======== outputs: hist [B,K,DEPTH] i32, then scores [B,K] f32 ========
  for (int i = tid; i < NK * NDEPTH; i += 512)
    outHist[b * NK * NDEPTH + i] = hist[i];
  if (tid < NK)
    outScores[b * NK + tid] = scores[tid];
}

extern "C" void kernel_launch(void* const* d_in, const int* in_sizes, int n_in,
                              void* d_out, int out_size, void* d_ws, size_t ws_size,
                              hipStream_t stream) {
  (void)in_sizes; (void)n_in; (void)out_size; (void)ws_size;
  // inputs: 0=env(unused by reference), 1=root_state, 2=W_logits, 3=b_logits,
  //         4=W_hh, 5=b_hh, 6=emb, 7=beams(=128)
  const float* root = (const float*)d_in[1];
  const float* Wl   = (const float*)d_in[2];
  const float* bl   = (const float*)d_in[3];
  const float* Whh  = (const float*)d_in[4];
  const float* bhh  = (const float*)d_in[5];
  const float* emb  = (const float*)d_in[6];

  // workspace: bf16 fragment-packed weights
  u16* wlFrag  = (u16*)d_ws;                          // 512*64  bf16 = 64 KB
  u16* whhFrag = (u16*)((char*)d_ws + 65536);         // 512*512 bf16 = 512 KB

  SubsetSelector_prep_kernel<<<(NH * ND + 255) / 256, 256, 0, stream>>>(
      Wl, wlFrag, ND, ND / 16, NH * ND);
  SubsetSelector_prep_kernel<<<(NH * NH + 255) / 256, 256, 0, stream>>>(
      Whh, whhFrag, NH, NH / 16, NH * NH);

  int*   outHist   = (int*)d_out;
  float* outScores = (float*)d_out + (size_t)NB * NK * NDEPTH;

  hipFuncSetAttribute((const void*)SubsetSelector_6889127543098_kernel,
                      hipFuncAttributeMaxDynamicSharedMemorySize, SMEM_BYTES);
  SubsetSelector_6889127543098_kernel<<<NB, 512, SMEM_BYTES, stream>>>(
      wlFrag, whhFrag, bl, bhh, emb, root, outHist, outScores);
}
